// OGBGGIN_41377714929868
// MI455X (gfx1250) — compile-verified, matrix operand order checked by reading, not run on hardware
//
#include <hip/hip_runtime.h>
#include <hip/hip_bf16.h>

// ---------------------------------------------------------------------------
// Problem constants (match reference)
// ---------------------------------------------------------------------------
#define NN     50000
#define EE     800000
#define GG     256
#define LL     3
#define DD     300
#define DP     304      // DD padded to a multiple of 16 (pad cols kept at 0)
#define DD2    600
#define NF_    128
#define EF_    16
#define OUT_   128
#define BN_EPS 1e-5f

typedef __bf16 v16bf __attribute__((ext_vector_type(16)));
typedef float  v8f   __attribute__((ext_vector_type(8)));

union Frag {            // one WMMA A/B operand: 16 bf16 = 8 VGPRs = 32 bytes
    v16bf v;
    uint4 q[2];
};

// ---------------------------------------------------------------------------
// Tiled bf16-WMMA GEMM:  C = op(A) @ B + bias  (+ optional column stats)
// Block: 256 threads = 8 waves. Block tile: 64(M) x 128(N), K step 32.
// Wave w -> M-subtile w/2, N-subtiles (w&1)*4 .. +3  (4 accumulators/wave).
// MODE 0: A as-is   MODE 1: A = A + A2   MODE 2: A = relu(A*scale + shift)
// sA / sC: row strides of A (and A2) / C, decoupled from K / Ncols.
// ---------------------------------------------------------------------------
#define BM 64
#define BN 128
#define BK 32

template <int MODE, bool STATS>
__global__ __launch_bounds__(256) void gemm_wmma_kernel(
    const float* __restrict__ A, const float* __restrict__ A2,
    const float* __restrict__ A3, const float* __restrict__ B,
    const float* __restrict__ bias, float* __restrict__ C,
    float* __restrict__ colsum, float* __restrict__ colsumsq,
    int M, int K, int Ncols, int sA, int sC)
{
    __shared__ __align__(16) __bf16 As[BM * BK];   // row-major [64][32]
    __shared__ __align__(16) __bf16 Bt[BN * BK];   // B^T tile  [128][32]

    const int t    = threadIdx.x;
    const int lane = t & 31;
    const int wave = t >> 5;
    const int half = lane >> 4;      // lanes 16..31
    const int r16  = lane & 15;
    const int msub = wave >> 1;
    const int ngrp = wave & 1;
    const int m0   = blockIdx.y * BM;
    const int n0   = blockIdx.x * BN;

    v8f acc[4];
#pragma unroll
    for (int j = 0; j < 4; ++j) { v8f z = {}; acc[j] = z; }

    // staging coordinates (fixed per thread)
    const int ar = t >> 2;            // A row within tile
    const int ak = (t & 3) * 8;       // A k-base within tile
    const int bn_ = t & 127;          // B column within tile
    const int bkq = t >> 7;           // 0..1

    const int numKt = (K + BK - 1) / BK;
    for (int kt = 0; kt < numKt; ++kt) {
        const int k0    = kt * BK;
        const bool fullK = (k0 + BK) <= K;

        // ---- stage A tile (64x32 f32 -> bf16), fusing MODE transform ----
        {
            const int row = m0 + ar;
            const float* Ap = A + row * sA + k0 + ak;
            if (row < M && fullK) {                        // interior fast path
                __builtin_prefetch(Ap + BK, 0, 1);         // next k-tile
#pragma unroll
                for (int i = 0; i < 8; ++i) {
                    float v = Ap[i];
                    if (MODE == 1)      v += A2[row * sA + k0 + ak + i];
                    else if (MODE == 2) v  = fmaxf(fmaf(v, A2[k0 + ak + i], A3[k0 + ak + i]), 0.f);
                    As[ar * BK + ak + i] = (__bf16)v;
                }
            } else {
#pragma unroll
                for (int i = 0; i < 8; ++i) {
                    const int k = k0 + ak + i;
                    float v = 0.f;
                    if (row < M && k < K) {
                        v = Ap[i];
                        if (MODE == 1)      v += A2[row * sA + k];
                        else if (MODE == 2) v  = fmaxf(fmaf(v, A2[k], A3[k]), 0.f);
                    }
                    As[ar * BK + ak + i] = (__bf16)v;
                }
            }
        }
        // ---- stage B^T tile (coalesced global read along N) ----
        {
            if (fullK && (n0 + BN) <= Ncols) {             // interior fast path
#pragma unroll
                for (int i = 0; i < 16; ++i) {
                    const int k = bkq * 16 + i;
                    Bt[bn_ * BK + k] = (__bf16)B[(k0 + k) * Ncols + n0 + bn_];
                }
            } else {
#pragma unroll
                for (int i = 0; i < 16; ++i) {
                    const int k  = bkq * 16 + i;
                    const int gk = k0 + k, gn = n0 + bn_;
                    float v = (gk < K && gn < Ncols) ? B[gk * Ncols + gn] : 0.f;
                    Bt[bn_ * BK + k] = (__bf16)v;
                }
            }
        }
        __syncthreads();

        // ---- fragments: ISA 16-bit layout => two contiguous 8xbf16 runs ----
        Frag a;
        const __bf16* ap = &As[(msub * 16 + r16) * BK + half * 8];
        a.q[0] = *(const uint4*)ap;
        a.q[1] = *(const uint4*)(ap + 16);
#pragma unroll
        for (int j = 0; j < 4; ++j) {
            Frag b;
            const __bf16* bp = &Bt[((ngrp * 4 + j) * 16 + r16) * BK + half * 8];
            b.q[0] = *(const uint4*)bp;
            b.q[1] = *(const uint4*)(bp + 16);
            acc[j] = __builtin_amdgcn_wmma_f32_16x16x32_bf16(
                false, a.v, false, b.v, (short)0, acc[j], false, false);
        }
        __syncthreads();
    }

    // ---- epilogue: bias, store, per-column sum / sum^2 ----
    const bool fullTile = ((m0 + BM) <= M) && ((n0 + BN) <= Ncols);
    const int  rowbase  = m0 + msub * 16 + half * 8;
#pragma unroll
    for (int j = 0; j < 4; ++j) {
        const int col = n0 + (ngrp * 4 + j) * 16 + r16;
        float s = 0.f, ss = 0.f;
        if (fullTile) {                                    // interior fast path
            const float bv = bias[col];
            float* Cp = C + rowbase * sC + col;
#pragma unroll
            for (int r = 0; r < 8; ++r) {
                const float v = acc[j][r] + bv;
                Cp[r * sC] = v;
                if (STATS) { s += v; ss += v * v; }
            }
        } else {
            const float bv = (col < Ncols) ? bias[col] : 0.f;
#pragma unroll
            for (int r = 0; r < 8; ++r) {
                const int row = rowbase + r;
                if (row < M && col < Ncols) {
                    const float v = acc[j][r] + bv;
                    C[row * sC + col] = v;
                    if (STATS) { s += v; ss += v * v; }
                }
            }
        }
        if (STATS) {
            s  += __shfl_xor(s,  16, 32);   // lanes L and L^16 share a column
            ss += __shfl_xor(ss, 16, 32);
            if (half == 0 && col < Ncols) {
                atomicAdd(&colsum[col],   s);
                atomicAdd(&colsumsq[col], ss);
            }
        }
    }
}

// ---------------------------------------------------------------------------
// Operand pre-pack kernels (run once per launch; invariant over edge blocks)
// ---------------------------------------------------------------------------
// edge_feat [E][16] f32 -> bf16 (A fragments become one b128 load)
__global__ void prep_ef_kernel(const float* __restrict__ ef,
                               __bf16* __restrict__ efb)
{
    int i = blockIdx.x * blockDim.x + threadIdx.x;
    if (i < EE * EF_) efb[i] = (__bf16)ef[i];
}

// edge_W [L][16][300] f32 -> eWt [L][304][16] bf16 (transposed, padded);
// edge_b [L][300]     f32 -> ebp [L][304]     f32 (padded)
__global__ void prep_ew_kernel(const float* __restrict__ eW,
                               const float* __restrict__ eb,
                               __bf16* __restrict__ eWt,
                               float* __restrict__ ebp)
{
    int i = blockIdx.x * blockDim.x + threadIdx.x;
    if (i < LL * DP * EF_) {
        int l   = i / (DP * EF_);
        int rem = i - l * DP * EF_;
        int col = rem >> 4;
        int k   = rem & 15;
        eWt[i] = (col < DD) ? (__bf16)eW[(l * EF_ + k) * DD + col] : (__bf16)0.f;
    }
    if (i < LL * DP) {
        int l = i / DP;
        int c = i - l * DP;
        ebp[i] = (c < DD) ? eb[l * DD + c] : 0.f;
    }
}

// ---------------------------------------------------------------------------
// Fused edge kernel: msg = relu(hn[src] + edge_feat @ eW + eb),
//                    agg[dst] += msg   (global f32 atomics)
// Block = 128 threads (4 waves) handles 16 edges; one WMMA per 16-col tile.
// All operands pre-packed/padded -> inner loop is fully guard-free:
// b128 frag loads -> v_wmma -> 8x (b32 gather + global_atomic_add_f32).
// ---------------------------------------------------------------------------
__global__ __launch_bounds__(128) void edge_fused_kernel(
    const __bf16* __restrict__ efb, const int* __restrict__ src,
    const int* __restrict__ dst, const float* __restrict__ hn,
    const __bf16* __restrict__ eWt, const float* __restrict__ ebp,
    float* __restrict__ agg)
{
    __shared__ int s_src[16];
    __shared__ int s_dst[16];

    const int t    = threadIdx.x;
    const int lane = t & 31;
    const int wave = t >> 5;
    const int half = lane >> 4;
    const int r16  = lane & 15;
    const int e0   = blockIdx.x * 16;

    if (t < 16)       s_src[t]      = src[e0 + t];
    else if (t < 32)  s_dst[t - 16] = dst[e0 + t - 16];
    __syncthreads();

    const uint4 z4 = make_uint4(0u, 0u, 0u, 0u);

    // A fragment: row = edge r16, valid K = half*8 + 0..7 (elements 0..7),
    // elements 8..15 map to K>=16 -> zero pad.  One 16B load, coalesced.
    Frag a;
    a.q[0] = *(const uint4*)(efb + (e0 + r16) * EF_ + half * 8);
    a.q[1] = z4;

    for (int ct = wave; ct < DP / 16; ct += 4) {          // 19 col tiles
        const int col = ct * 16 + r16;

        // B fragment: one 16B load from transposed/padded eWt, coalesced.
        Frag b;
        b.q[0] = *(const uint4*)(eWt + col * EF_ + half * 8);
        b.q[1] = z4;

        v8f c = {};
        c = __builtin_amdgcn_wmma_f32_16x16x32_bf16(
            false, a.v, false, b.v, (short)0, c, false, false);

        const float bcol = ebp[col];
#pragma unroll
        for (int r = 0; r < 8; ++r) {
            const int m  = half * 8 + r;                  // C row = edge in tile
            const int sv = s_src[m];
            const int dv = s_dst[m];
            // pad cols: b==0, bcol==0, hn pad==0 -> msg==0, harmless atomic
            float msg = fmaxf(c[r] + bcol + hn[sv * DP + col], 0.f);
            atomicAdd(&agg[dv * DP + col], msg);
        }
    }
}

// ---------------------------------------------------------------------------
// Small helper kernels
// ---------------------------------------------------------------------------
__global__ void zero_kernel(float* p, int n) {
    int i = blockIdx.x * blockDim.x + threadIdx.x;
    if (i < n) p[i] = 0.f;
}

__global__ void bn_finalize_kernel(const float* __restrict__ colsum,
                                   const float* __restrict__ colsumsq,
                                   const float* __restrict__ g,
                                   const float* __restrict__ beta,
                                   float* __restrict__ scale,
                                   float* __restrict__ shift,
                                   int C, float invM)
{
    int c = blockIdx.x * blockDim.x + threadIdx.x;
    if (c < C) {
        float m   = colsum[c] * invM;
        float var = colsumsq[c] * invM - m * m;
        float s   = g[c] * rsqrtf(var + BN_EPS);
        scale[c] = s;
        shift[c] = beta[c] - m * s;
    }
}

// hn[node*DP + c] = (relu?)(x2[node*DD + c]*scale[c] + shift[c]);  pads -> 0
__global__ void apply_bn_pad_kernel(const float* __restrict__ x,
                                    const float* __restrict__ scale,
                                    const float* __restrict__ shift,
                                    float* __restrict__ out, int do_relu)
{
    int i = blockIdx.x * blockDim.x + threadIdx.x;
    if (i < NN * DP) {
        int node = i / DP;
        int c    = i - node * DP;
        float v  = 0.f;
        if (c < DD) {
            v = fmaf(x[node * DD + c], scale[c], shift[c]);
            if (do_relu) v = fmaxf(v, 0.f);
        }
        out[i] = v;
    }
}

__global__ void pool_count_kernel(const int* __restrict__ gid, float* cnt) {
    int i = blockIdx.x * blockDim.x + threadIdx.x;
    if (i < NN) atomicAdd(&cnt[gid[i]], 1.f);
}

__global__ void pool_sum_kernel(const float* __restrict__ hn,
                                const int* __restrict__ gid,
                                float* __restrict__ hg)
{
    int i = blockIdx.x * blockDim.x + threadIdx.x;
    if (i < NN * DD) {
        int node = i / DD;
        int c    = i - node * DD;
        atomicAdd(&hg[gid[node] * DD + c], hn[node * DP + c]);
    }
}

__global__ void head_kernel(const float* __restrict__ hg,
                            const float* __restrict__ cnt,
                            const float* __restrict__ W,
                            const float* __restrict__ b,
                            float* __restrict__ out)
{
    int i = blockIdx.x * blockDim.x + threadIdx.x;
    if (i < GG * OUT_) {
        int g = i / OUT_;
        int o = i - g * OUT_;
        float inv = 1.f / fmaxf(cnt[g], 1.f);
        float acc = 0.f;
        for (int d = 0; d < DD; ++d)
            acc = fmaf(hg[g * DD + d], W[d * OUT_ + o], acc);
        out[i] = fmaf(acc, inv, b[o]);
    }
}

// ---------------------------------------------------------------------------
// Host-side orchestration
// ---------------------------------------------------------------------------
static inline int ceil_div(int a, int b) { return (a + b - 1) / b; }

extern "C" void kernel_launch(void* const* d_in, const int* in_sizes, int n_in,
                              void* d_out, int out_size, void* d_ws, size_t ws_size,
                              hipStream_t stream)
{
    const float* node_feat = (const float*)d_in[0];
    const float* edge_feat = (const float*)d_in[1];
    const int*   src       = (const int*)d_in[2];
    const int*   dst       = (const int*)d_in[3];
    const int*   gid       = (const int*)d_in[4];
    const float* node_W    = (const float*)d_in[6];
    const float* node_b    = (const float*)d_in[7];
    const float* edge_W    = (const float*)d_in[8];
    const float* edge_b    = (const float*)d_in[9];
    const float* W1        = (const float*)d_in[10];
    const float* b1        = (const float*)d_in[11];
    const float* g1        = (const float*)d_in[12];
    const float* beta1     = (const float*)d_in[13];
    const float* W2        = (const float*)d_in[14];
    const float* b2        = (const float*)d_in[15];
    const float* g2        = (const float*)d_in[16];
    const float* beta2     = (const float*)d_in[17];
    const float* pred_W    = (const float*)d_in[18];
    const float* pred_b    = (const float*)d_in[19];

    // workspace layout (~328 MB)
    float* ws      = (float*)d_ws;
    float* hn      = ws;                       // N*DP  = 15,200,000 (padded)
    float* agg     = hn + NN * DP;             // N*DP            (padded)
    float* x1      = agg + NN * DP;            // N*2D  = 30,000,000
    float* x2      = x1 + NN * DD2;            // N*D   = 15,000,000
    float* colsum1 = x2 + NN * DD;             // 600
    float* colsq1  = colsum1 + DD2;            // 600
    float* scale1  = colsq1 + DD2;             // 600
    float* shift1  = scale1 + DD2;             // 600
    float* colsum2 = shift1 + DD2;             // 300
    float* colsq2  = colsum2 + DD;             // 300
    float* scale2  = colsq2 + DD;              // 300
    float* shift2  = scale2 + DD;              // 300
    float* hg      = shift2 + DD;              // G*D = 76,800
    float* cnt     = hg + GG * DD;             // 256
    float* ebp     = cnt + GG;                 // L*DP = 912 (padded edge bias)
    __bf16* efb    = (__bf16*)(ebp + LL * DP); // E*16 bf16 = 25.6 MB
    __bf16* eWt    = efb + (size_t)EE * EF_;   // L*DP*16 bf16 (transposed, padded)

    const dim3 thr256(256);
    const int mBlocks = ceil_div(NN, BM);      // 782

    // ---- pre-pack edge operands (bf16 / transposed / padded) ----
    prep_ef_kernel<<<ceil_div(EE * EF_, 256), thr256, 0, stream>>>(edge_feat, efb);
    prep_ew_kernel<<<ceil_div(LL * DP * EF_, 256), thr256, 0, stream>>>(
        edge_W, edge_b, eWt, ebp);

    // zero hn (incl. pad cols) so gemm0's unwritten pads are 0
    zero_kernel<<<ceil_div(NN * DP, 256), thr256, 0, stream>>>(hn, NN * DP);

    // node encoder: hn = node_feat @ node_W + node_b  (hn stride DP)
    gemm_wmma_kernel<0, false><<<dim3(ceil_div(DD, BN), mBlocks), thr256, 0, stream>>>(
        node_feat, nullptr, nullptr, node_W, node_b, hn, nullptr, nullptr,
        NN, NF_, DD, NF_, DP);

    for (int l = 0; l < LL; ++l) {
        zero_kernel<<<ceil_div(NN * DP, 256), thr256, 0, stream>>>(agg, NN * DP);
        zero_kernel<<<ceil_div(2 * DD2, 256), thr256, 0, stream>>>(colsum1, 2 * DD2);
        zero_kernel<<<ceil_div(2 * DD, 256), thr256, 0, stream>>>(colsum2, 2 * DD);

        // fused edge encoder + gather + relu + scatter-add
        edge_fused_kernel<<<EE / 16, dim3(128), 0, stream>>>(
            efb, src, dst, hn, eWt + l * DP * EF_, ebp + l * DP, agg);

        // x1 = (hn + agg) @ W1 + b1, with column stats
        gemm_wmma_kernel<1, true><<<dim3(ceil_div(DD2, BN), mBlocks), thr256, 0, stream>>>(
            hn, agg, nullptr, W1 + l * DD * DD2, b1 + l * DD2, x1, colsum1, colsq1,
            NN, DD, DD2, DP, DD2);
        bn_finalize_kernel<<<ceil_div(DD2, 256), thr256, 0, stream>>>(
            colsum1, colsq1, g1 + l * DD2, beta1 + l * DD2, scale1, shift1,
            DD2, 1.f / (float)NN);

        // x2 = relu(BN1(x1)) @ W2 + b2, with column stats (BN+relu fused in A load)
        gemm_wmma_kernel<2, true><<<dim3(ceil_div(DD, BN), mBlocks), thr256, 0, stream>>>(
            x1, scale1, shift1, W2 + l * DD2 * DD, b2 + l * DD, x2, colsum2, colsq2,
            NN, DD2, DD, DD2, DD);
        bn_finalize_kernel<<<ceil_div(DD, 256), thr256, 0, stream>>>(
            colsum2, colsq2, g2 + l * DD, beta2 + l * DD, scale2, shift2,
            DD, 1.f / (float)NN);

        // hn = (relu?)(BN2(x2)), pads written to 0
        apply_bn_pad_kernel<<<ceil_div(NN * DP, 256), thr256, 0, stream>>>(
            x2, scale2, shift2, hn, (l != LL - 1) ? 1 : 0);
    }

    // mean pool per graph + prediction head
    zero_kernel<<<ceil_div(GG * DD + GG, 256), thr256, 0, stream>>>(hg, GG * DD + GG);
    pool_count_kernel<<<ceil_div(NN, 256), thr256, 0, stream>>>(gid, cnt);
    pool_sum_kernel<<<ceil_div(NN * DD, 256), thr256, 0, stream>>>(hn, gid, hg);
    head_kernel<<<ceil_div(GG * OUT_, 256), thr256, 0, stream>>>(
        hg, cnt, pred_W, pred_b, (float*)d_out);
}
